// BinPackingLSTMModel_39565238731449
// MI455X (gfx1250) — matrix operand order repaired and checked by reading, not compiled, not run on hardware
//
#include <hip/hip_runtime.h>
#include <hip/hip_fp16.h>

// Problem sizes (fixed by the reference)
#define BSZ 512
#define ISZ 256
#define HSZ 1024
#define OSZ 128
#define TSZ 100
#define GSZ (4 * HSZ)  // 4096 gate columns
#define HKT (HSZ / 32) // k-tiles when K == H

typedef __attribute__((ext_vector_type(16))) _Float16 v16h;
typedef __attribute__((ext_vector_type(8)))  float    v8f;

// ---------------------------------------------------------------------------
// WMMA fragment packing (wave32, v_wmma_f32_16x16x32_f16):
//  A (16x32, MxK), ISA 7.12.2: lane m (m<16) holds row M=m, K={0..7,16..23};
//                              lane m+16 holds row M=m, K={8..15,24..31}
//  B (32x16, KxN): lane n (n<16) holds col N=n, K={0..7,16..23}; lane n+16 holds
//                  col N=n, K={8..15,24..31}
//  C/D (16x16 f32): VGPR v, lanes 0-15: M=v, N=lane; lanes 16-31: M=v+8, N=lane-16
// Packed buffers: [Mtiles|Ntiles][Ktiles][32 lanes][16 halves] -> each lane's
// fragment is one contiguous 32-byte (2 x b128) load.
// ---------------------------------------------------------------------------

__device__ __forceinline__ size_t a_pack_idx(int m_g, int k, int KT) {
  int mt = m_g >> 4, m = m_g & 15, kt = k >> 5, kk = k & 31;
  int hi   = (kk >> 3) & 1;
  int lane = m + (hi << 4);
  int j    = (kk & 7) + ((kk >> 4) << 3);
  return (((size_t)(mt * KT + kt)) * 32 + lane) * 16 + j;
}

__device__ __forceinline__ size_t b_pack_idx(int n_g, int k, int KT) {
  int nt = n_g >> 4, kt = k >> 5;
  int lane = (n_g & 15) + (((k >> 3) & 1) << 4);
  int j    = (k & 7) + (((k >> 4) & 1) << 3);
  return (((size_t)(nt * KT + kt)) * 32 + lane) * 16 + j;
}

__global__ void pack_a_f16(const float* __restrict__ src, _Float16* __restrict__ dst,
                           int M, int K) {
  int idx = blockIdx.x * blockDim.x + threadIdx.x;
  if (idx >= M * K) return;
  int m = idx / K, k = idx % K;
  dst[a_pack_idx(m, k, K >> 5)] = (_Float16)src[idx];
}

__global__ void pack_b_f16(const float* __restrict__ w1, const float* __restrict__ w2,
                           _Float16* __restrict__ dst, int N, int K) {
  int idx = blockIdx.x * blockDim.x + threadIdx.x;
  if (idx >= N * K) return;
  int n = idx / K, k = idx % K;
  float v = w1[idx];
  if (w2) v += w2[idx];
  dst[b_pack_idx(n, k, K >> 5)] = (_Float16)v;
}

__global__ void vec_add_f32(const float* __restrict__ a, const float* __restrict__ b,
                            float* __restrict__ o, int n) {
  int i = blockIdx.x * blockDim.x + threadIdx.x;
  if (i < n) o[i] = a[i] + b[i];
}

__device__ __forceinline__ float sigm(float x) { return 1.0f / (1.0f + __expf(-x)); }

// ---------------------------------------------------------------------------
// Fused GEMM + LSTM cell step.
// Wave -> one 16-row batch tile x the SAME 16 hidden columns in all 4 gates:
//   ntile(g) = g * (H/16) + hct ,  g in {i,f,g,o}
// After the k-loop each wave holds i/f/g/o for a 16x16 (batch x hidden) patch,
// applies the cell non-linearities, updates c (f32) and writes the new h
// directly into the packed A-fragment buffer feeding the next step's GEMMs.
// Grid: x = H/16 (=64 hidden-column tiles), y = B / (16*wavesPerBlock).
// ---------------------------------------------------------------------------
__global__ void wmma_lstm_step(const _Float16* __restrict__ pA,
                               const _Float16* __restrict__ pB,
                               const float* __restrict__ bias,
                               float* __restrict__ cst,
                               _Float16* __restrict__ hOut,
                               int KT, int zeroC) {
  const int lane  = threadIdx.x & 31;
  const int wave  = threadIdx.x >> 5;
  const int wpb   = blockDim.x >> 5;
  const int mtile = blockIdx.y * wpb + wave;
  const int hct   = blockIdx.x;  // hidden-column tile 0..63

  v8f acc[4] = {v8f{}, v8f{}, v8f{}, v8f{}};

  const _Float16* aBase = pA + ((size_t)mtile * KT) * 32 * 16;
  for (int kt = 0; kt < KT; ++kt) {
    // Batch all fragment loads first, then run the WMMA chain back-to-back.
    v16h a = *(const v16h*)(aBase + ((size_t)kt * 32 + lane) * 16);
    v16h b[4];
#pragma unroll
    for (int g = 0; g < 4; ++g) {
      const int ntile = g * (HSZ / 16) + hct;
      b[g] = *(const v16h*)(pB + (((size_t)ntile * KT + kt) * 32 + lane) * 16);
    }
    if (kt + 1 < KT)
      __builtin_prefetch(aBase + ((size_t)(kt + 1) * 32 + lane) * 16, 0, 1);
#pragma unroll
    for (int g = 0; g < 4; ++g)
      acc[g] = __builtin_amdgcn_wmma_f32_16x16x32_f16(
          false, a, false, b[g], (short)0, acc[g], false, false);
  }

  // --- fused LSTM epilogue ---
  const int l      = lane & 15;
  const int rowOff = (lane < 16) ? 0 : 8;
  const int hcol   = hct * 16 + l;
  const float bi = bias[hcol];
  const float bf = bias[HSZ + hcol];
  const float bg = bias[2 * HSZ + hcol];
  const float bo = bias[3 * HSZ + hcol];

  union { v8f v; float f[8]; } ui, uf, ug, uo;
  ui.v = acc[0]; uf.v = acc[1]; ug.v = acc[2]; uo.v = acc[3];

#pragma unroll
  for (int v = 0; v < 8; ++v) {
    const int row = mtile * 16 + rowOff + v;
    const float ig = ui.f[v] + bi;
    const float fg = uf.f[v] + bf;
    const float gg = ug.f[v] + bg;
    const float og = uo.f[v] + bo;
    const float c  = zeroC ? 0.0f : cst[(size_t)row * HSZ + hcol];
    const float cn = sigm(fg) * c + sigm(ig) * tanhf(gg);
    const float hn = sigm(og) * tanhf(cn);
    cst[(size_t)row * HSZ + hcol] = cn;
    hOut[a_pack_idx(row, hcol, HKT)] = (_Float16)hn;
  }
}

// ---------------------------------------------------------------------------
// Plain GEMM + bias (used for the output FC): out[m,n] = sum_k A[m,k]W[n,k] + b[n]
// ---------------------------------------------------------------------------
__global__ void wmma_gemm_bias(const _Float16* __restrict__ pA,
                               const _Float16* __restrict__ pB,
                               const float* __restrict__ bias,
                               float* __restrict__ out,
                               int KT, int ldOut) {
  const int lane   = threadIdx.x & 31;
  const int wave   = threadIdx.x >> 5;
  const int wpb    = blockDim.x >> 5;
  const int mtile  = blockIdx.y * wpb + wave;
  const int ntile0 = blockIdx.x * 4;

  v8f acc[4] = {v8f{}, v8f{}, v8f{}, v8f{}};
  const _Float16* aBase = pA + ((size_t)mtile * KT) * 32 * 16;
  for (int kt = 0; kt < KT; ++kt) {
    v16h a = *(const v16h*)(aBase + ((size_t)kt * 32 + lane) * 16);
    v16h b[4];
#pragma unroll
    for (int nt = 0; nt < 4; ++nt)
      b[nt] = *(const v16h*)(pB + (((size_t)(ntile0 + nt) * KT + kt) * 32 + lane) * 16);
    if (kt + 1 < KT)
      __builtin_prefetch(aBase + ((size_t)(kt + 1) * 32 + lane) * 16, 0, 1);
#pragma unroll
    for (int nt = 0; nt < 4; ++nt)
      acc[nt] = __builtin_amdgcn_wmma_f32_16x16x32_f16(
          false, a, false, b[nt], (short)0, acc[nt], false, false);
  }

  const int l      = lane & 15;
  const int rowOff = (lane < 16) ? 0 : 8;
#pragma unroll
  for (int nt = 0; nt < 4; ++nt) {
    int col  = (ntile0 + nt) * 16 + l;
    float bv = bias[col];
    union { v8f v; float f[8]; } u;
    u.v = acc[nt];
#pragma unroll
    for (int v = 0; v < 8; ++v)
      out[(size_t)(mtile * 16 + rowOff + v) * ldOut + col] = u.f[v] + bv;
  }
}

// ---------------------------------------------------------------------------
extern "C" void kernel_launch(void* const* d_in, const int* in_sizes, int n_in,
                              void* d_out, int out_size, void* d_ws, size_t ws_size,
                              hipStream_t stream) {
  (void)in_sizes; (void)n_in; (void)out_size; (void)ws_size;
  const float* x       = (const float*)d_in[0];
  const float* enc_Wih = (const float*)d_in[1];
  // d_in[2] = enc_Whh: multiplied by h==0 in the reference -> unused
  const float* enc_bih = (const float*)d_in[3];
  const float* enc_bhh = (const float*)d_in[4];
  const float* dec_Wih = (const float*)d_in[5];
  const float* dec_Whh = (const float*)d_in[6];
  const float* dec_bih = (const float*)d_in[7];
  const float* dec_bhh = (const float*)d_in[8];
  const float* fc_W    = (const float*)d_in[9];
  const float* fc_b    = (const float*)d_in[10];
  float* out = (float*)d_out;

  // Workspace carve-out (~22.5 MB), 256B aligned slices.
  char* ws = (char*)d_ws;
  size_t off = 0;
  auto carve = [&](size_t bytes) -> void* {
    off = (off + 255) & ~(size_t)255;
    void* p = (void*)(ws + off);
    off += bytes;
    return p;
  };
  _Float16* pBenc  = (_Float16*)carve((size_t)GSZ * ISZ * 2);  // enc_Wih packed
  _Float16* pBdec0 = (_Float16*)carve((size_t)GSZ * HSZ * 2);  // dec_Whh packed (t=0)
  _Float16* pBdec  = (_Float16*)carve((size_t)GSZ * HSZ * 2);  // dec_Wih+dec_Whh packed
  _Float16* pBfc   = (_Float16*)carve((size_t)OSZ * HSZ * 2);  // fc_W packed
  _Float16* pX     = (_Float16*)carve((size_t)BSZ * ISZ * 2);  // x packed (A frags)
  _Float16* pH0    = (_Float16*)carve((size_t)BSZ * HSZ * 2);  // h ping
  _Float16* pH1    = (_Float16*)carve((size_t)BSZ * HSZ * 2);  // h pong
  float*    cst    = (float*)carve((size_t)BSZ * HSZ * 4);
  float*    benc   = (float*)carve((size_t)GSZ * 4);
  float*    bdec   = (float*)carve((size_t)GSZ * 4);
  _Float16* pH[2]  = {pH0, pH1};

  const int TB = 256;
  // --- one-time prep ---
  pack_b_f16<<<(GSZ * ISZ + TB - 1) / TB, TB, 0, stream>>>(enc_Wih, nullptr, pBenc, GSZ, ISZ);
  pack_b_f16<<<(GSZ * HSZ + TB - 1) / TB, TB, 0, stream>>>(dec_Whh, nullptr, pBdec0, GSZ, HSZ);
  pack_b_f16<<<(GSZ * HSZ + TB - 1) / TB, TB, 0, stream>>>(dec_Wih, dec_Whh, pBdec, GSZ, HSZ);
  pack_b_f16<<<(OSZ * HSZ + TB - 1) / TB, TB, 0, stream>>>(fc_W, nullptr, pBfc, OSZ, HSZ);
  pack_a_f16<<<(BSZ * ISZ + TB - 1) / TB, TB, 0, stream>>>(x, pX, BSZ, ISZ);
  vec_add_f32<<<(GSZ + TB - 1) / TB, TB, 0, stream>>>(enc_bih, enc_bhh, benc, GSZ);
  vec_add_f32<<<(GSZ + TB - 1) / TB, TB, 0, stream>>>(dec_bih, dec_bhh, bdec, GSZ);

  dim3 gStep(HSZ / 16, BSZ / 128);  // 64 x 4 blocks of 8 waves
  dim3 gFc(OSZ / 64, BSZ / 16);     // 2 x 32 blocks of 1 wave

  // --- encoder: one fused cell step from zero state (gates = x@Wih.T + b) ---
  wmma_lstm_step<<<gStep, TB, 0, stream>>>(pX, pBenc, benc, cst, pH[0],
                                           ISZ / 32, /*zeroC=*/1);

  // --- decoder: t=0 uses Whh only (dec_in==0); t>=1 uses Wih+Whh (dec_in==h).
  //     pH ping-pongs so no block reads a buffer another block is writing. ---
  for (int t = 0; t < TSZ; ++t) {
    wmma_lstm_step<<<gStep, TB, 0, stream>>>(pH[t & 1], (t == 0) ? pBdec0 : pBdec,
                                             bdec, cst, pH[(t + 1) & 1],
                                             HKT, /*zeroC=*/0);
    // out[b, t, :] = h_t @ fc_W.T + fc_b
    wmma_gemm_bias<<<gFc, 32, 0, stream>>>(pH[(t + 1) & 1], pBfc, fc_b,
                                           out + (size_t)t * OSZ, HKT, TSZ * OSZ);
  }
}
